// MultiLoRALayerSTK_45535243272923
// MI455X (gfx1250) — matrix operand order repaired and checked by reading, not compiled, not run on hardware
//
#include <hip/hip_runtime.h>

typedef __attribute__((ext_vector_type(16))) _Float16 v16h;
typedef __attribute__((ext_vector_type(8)))  _Float16 v8h;
typedef __attribute__((ext_vector_type(4)))  _Float16 v4h;
typedef __attribute__((ext_vector_type(8)))  float    v8f;

#define BATCH 4
#define SEQ   4096
#define HID   4096
#define OUTF  4096
#define NADP  8
#define RANK  64
#define BSTR  (NADP * RANK)     // 512 floats between h-rows of B
#define SCALE (1.0f / 64.0f)

#define BM  64                  // rows of x per workgroup
#define KB  32                  // K-slab per GEMM-1 iteration (WMMA K)

// ---- WMMA fragment layouts (ISA 7.12.2, wave32) ----
// A 16x32 f16: element (m,K) -> lane (m&15)|(((K>>3)&1)<<4), slot ((K>>4)<<3)|(K&7)
// B 32x16 f16: element (K,n) -> lane (n&15)|(((K>>3)&1)<<4), slot ((K>>4)<<3)|(K&7)
// C 16x16 f32: reg r holds row r+8*(lane>=16), col lane&15

__global__ __launch_bounds__(256)
void lora_fused_wmma(const float* __restrict__ x, const int* __restrict__ ids,
                     const float* __restrict__ A, const float* __restrict__ Bm,
                     float* __restrict__ out)
{
    // Fragment-ordered LDS tiles, double-buffered for stage 1
    __shared__ __align__(16) _Float16 lxf[2][4 * 32 * 16];  // x  A-frags
    __shared__ __align__(16) _Float16 lbf[2][4 * 32 * 16];  // B  B-frags
    __shared__ __align__(16) _Float16 lbxf[4 * 32 * 32];    // Bx A-frags (K=64)

    const int b    = blockIdx.y;
    const int row0 = blockIdx.x * BM;
    const int tid  = threadIdx.x;
    const int wave = tid >> 5;
    const int lane = tid & 31;
    const int a    = ids[b];

    const float* xb   = x  + ((size_t)b * SEQ + row0) * HID;
    const float* Bsel = Bm + (size_t)a * RANK;              // + h*BSTR + r
    const float* Asel = A  + (size_t)a * RANK * OUTF;       // + k*OUTF + n

    // wave w owns mtile = w>>1, ntiles {2*(w&1), 2*(w&1)+1}
    const int mt  = wave >> 1;
    const int nt0 = (wave & 1) * 2;
    v8f acc0 = {}; v8f acc1 = {};

    // --- per-thread staging coordinates (compile-time trip counts) ---
    // x: two float4 per thread
    const int xr0   = tid >> 3;                 // rows 0..31
    const int xr1   = xr0 + 32;                 // rows 32..63
    const int xc    = (tid & 7) << 2;           // K base of the float4
    const int xslot = ((xc >> 4) << 3) | (xc & 7);
    const int xlh   = ((xc >> 3) & 1) << 4;
    const int xdst0 = ((((xr0 >> 4) << 5) | (xr0 & 15) | xlh) << 4) + xslot;
    const int xdst1 = ((((xr1 >> 4) << 5) | (xr1 & 15) | xlh) << 4) + xslot;
    // B: 8 K-rows of one column per thread
    const int bcol = tid & 63;
    const int bkh  = tid >> 6;
    const int bdst = ((((bcol >> 4) << 5) | (bcol & 15) | ((bkh & 1) << 4)) << 4)
                   + (bkh >> 1) * 8;

    // ---- Stage 1: software-pipelined K loop, loads in flight during WMMA ----
    float4 gx0, gx1;
    float  gb[8];
    {
        gx0 = *(const float4*)(xb + (size_t)xr0 * HID + xc);
        gx1 = *(const float4*)(xb + (size_t)xr1 * HID + xc);
        const float* bp = Bsel + (size_t)(bkh * 8) * BSTR + bcol;
        #pragma unroll
        for (int j = 0; j < 8; ++j) gb[j] = bp[(size_t)j * BSTR];
    }

    const int NIT = HID / KB;
    for (int it = 0; it < NIT; ++it) {
        const int buf = it & 1;
        // convert the in-flight slab and deposit in fragment order
        {
            v4h h;
            h[0] = (_Float16)gx0.x; h[1] = (_Float16)gx0.y;
            h[2] = (_Float16)gx0.z; h[3] = (_Float16)gx0.w;
            *(v4h*)&lxf[buf][xdst0] = h;
            h[0] = (_Float16)gx1.x; h[1] = (_Float16)gx1.y;
            h[2] = (_Float16)gx1.z; h[3] = (_Float16)gx1.w;
            *(v4h*)&lxf[buf][xdst1] = h;
            v8h hb;
            #pragma unroll
            for (int j = 0; j < 8; ++j) hb[j] = (_Float16)gb[j];
            *(v8h*)&lbf[buf][bdst] = hb;
        }
        __syncthreads();
        // issue next slab's global loads; they fly during this slab's WMMAs
        if (it + 1 < NIT) {
            const int k0 = (it + 1) * KB;
            gx0 = *(const float4*)(xb + (size_t)xr0 * HID + k0 + xc);
            gx1 = *(const float4*)(xb + (size_t)xr1 * HID + k0 + xc);
            const float* bp = Bsel + (size_t)(k0 + bkh * 8) * BSTR + bcol;
            #pragma unroll
            for (int j = 0; j < 8; ++j) gb[j] = bp[(size_t)j * BSTR];
        }
        const v16h af  = *(const v16h*)&lxf[buf][((mt << 5) | lane) << 4];
        const v16h bf0 = *(const v16h*)&lbf[buf][((nt0 << 5) | lane) << 4];
        const v16h bf1 = *(const v16h*)&lbf[buf][(((nt0 + 1) << 5) | lane) << 4];
        acc0 = __builtin_amdgcn_wmma_f32_16x16x32_f16(false, af, false, bf0,
                                                      (short)0, acc0, false, false);
        acc1 = __builtin_amdgcn_wmma_f32_16x16x32_f16(false, af, false, bf1,
                                                      (short)0, acc1, false, false);
    }

    // ---- deposit Bx into LDS in stage-2 A-fragment order ----
    {
        const int rowb = mt * 16 + ((lane >> 4) << 3);
        const int K0   = nt0 * 16 + (lane & 15);
        #pragma unroll
        for (int p = 0; p < 2; ++p) {
            const int K    = K0 + p * 16;
            const int s    = K >> 5;
            const int Kloc = K & 31;
            const int slot = (s << 4) | (Kloc & 15);
            const int lnH  = (Kloc >> 4) << 4;
            #pragma unroll
            for (int r = 0; r < 8; ++r) {
                const int row = rowb + r;
                const float v = p ? acc1[r] : acc0[r];
                lbxf[(((row >> 4) << 5) | (row & 15) | lnH) * 32 + slot] = (_Float16)v;
            }
        }
    }
    __syncthreads();

    // ---- Stage 2: out(64x4096) = Bx @ A_sel * 1/64 ----
    v16h a2[4][2];                                   // 4 mtiles x 2 K-slabs
    #pragma unroll
    for (int m = 0; m < 4; ++m)
        #pragma unroll
        for (int s = 0; s < 2; ++s)
            a2[m][s] = *(const v16h*)&lbxf[(((m << 5) | lane) << 5) + (s << 4)];

    float* outb = out + ((size_t)b * SEQ + row0) * OUTF;
    const int kbase = (lane >> 4) << 4;              // B-frag K base for this lane-half
    const int colw  = wave * (OUTF / 8);             // 512-column strip per wave
    for (int nt = 0; nt < (OUTF / 8) / 16; ++nt) {
        const int col = colw + nt * 16 + (lane & 15);
        v16h bfA, bfB;                                // A_sel as B-operand, K=0..31 / 32..63
        #pragma unroll
        for (int e = 0; e < 16; ++e) {
            bfA[e] = (_Float16)Asel[(size_t)(kbase + e) * OUTF + col];
            bfB[e] = (_Float16)Asel[(size_t)(kbase + e + 32) * OUTF + col];
        }
        #pragma unroll
        for (int m = 0; m < 4; ++m) {
            v8f acc = {};
            acc = __builtin_amdgcn_wmma_f32_16x16x32_f16(false, a2[m][0], false, bfA,
                                                         (short)0, acc, false, false);
            acc = __builtin_amdgcn_wmma_f32_16x16x32_f16(false, a2[m][1], false, bfB,
                                                         (short)0, acc, false, false);
            const int rbase = m * 16 + ((lane >> 4) << 3);
            #pragma unroll
            for (int r = 0; r < 8; ++r)
                outb[(size_t)(rbase + r) * OUTF + col] = acc[r] * SCALE;
        }
    }
}

extern "C" void kernel_launch(void* const* d_in, const int* in_sizes, int n_in,
                              void* d_out, int out_size, void* d_ws, size_t ws_size,
                              hipStream_t stream) {
    const float* x   = (const float*)d_in[0];
    const int*   ids = (const int*)d_in[1];
    const float* A   = (const float*)d_in[2];
    const float* Bm  = (const float*)d_in[3];
    float*       out = (float*)d_out;

    dim3 grid(SEQ / BM, BATCH);
    lora_fused_wmma<<<grid, 256, 0, stream>>>(x, ids, A, Bm, out);
}
